// TransformerBlock_16535624090296
// MI455X (gfx1250) — matrix operand order repaired and checked by reading, not compiled, not run on hardware
//
#include <hip/hip_runtime.h>
#include <hip/hip_bf16.h>

// ---------------------------------------------------------------------------
// MI455X (gfx1250) transformer block, bf16 WMMA everywhere:
//   scores = QK^T/(sqrt(D)+eps); attn = softmax; X = Q + attn@V
//   out = relu(X@W1^T + b1) @ W2^T + b2 + X
// ---------------------------------------------------------------------------

typedef __attribute__((ext_vector_type(16))) __bf16 v16bf;
typedef __attribute__((ext_vector_type(8)))  float  v8f;

// pointer types for the gfx1250 async-to-LDS builtin (signature learned from
// clang diagnostic: param0 = vector_size(16) int, address_space(1))
typedef __attribute__((__vector_size__(16))) int vi4_t;
typedef vi4_t __attribute__((address_space(1)))* gvi4_p;   // global
typedef vi4_t __attribute__((address_space(3)))* lvi4_p;   // LDS

#define B_   16
#define S_   2048
#define D_   1024
#define NW   8        // waves per block (wave32 -> 256 threads)
#define CK   32       // key-chunk (WMMA K dim for bf16)
#define DSL  128      // per-wave D slice in attention (D_/NW)
#define PST  40       // P-tile row stride in bf16 elems (80B, 16B-aligned)
#define AST  1032     // FFN LDS A-panel row stride in bf16 elems (2064B)

#if __has_builtin(__builtin_amdgcn_global_load_async_to_lds_b128) && \
    __has_builtin(__builtin_amdgcn_s_wait_asynccnt)
#define USE_ASYNC_LDS 1
#else
#define USE_ASYNC_LDS 0
#endif

// scheduling fence: keep operand-load batches ahead of the WMMA burst so the
// loads clause together and their s_wait_loadcnt's stagger (no per-WMMA stall)
#if __has_builtin(__builtin_amdgcn_sched_barrier)
#define SCHED_FENCE() __builtin_amdgcn_sched_barrier(0)
#else
#define SCHED_FENCE()
#endif

static __device__ __forceinline__ unsigned short f2bf(float f) {
  unsigned u = __float_as_uint(f);
  u += 0x7FFFu + ((u >> 16) & 1u);          // round-to-nearest-even
  return (unsigned short)(u >> 16);
}

// A-operand (16-bit 16x32, ISA 7.12.2): lane reads elems [hi*8..+7] and [hi*8+16..+7]
static __device__ __forceinline__ v16bf loadA16(const unsigned short* p, int hi) {
  const uint4* q = reinterpret_cast<const uint4*>(p) + hi;   // +hi*16B
  union { uint4 u[2]; v16bf v; } t;
  t.u[0] = q[0];            // elems  0.. 7 (rel)
  t.u[1] = q[2];            // elems 16..23 (rel)
  return t.v;
}

// B-operand (32x16 from transposed row-major storage): lane reads elems [hi*8..+7] and [hi*8+16..+7]
static __device__ __forceinline__ v16bf loadB16(const unsigned short* p, int hi) {
  const uint4* q = reinterpret_cast<const uint4*>(p) + hi;
  union { uint4 u[2]; v16bf v; } t;
  t.u[0] = q[0];
  t.u[1] = q[2];
  return t.v;
}

static __device__ __forceinline__ v8f wmma_bf16(v16bf a, v16bf b, v8f c) {
  return __builtin_amdgcn_wmma_f32_16x16x32_bf16(false, a, false, b,
                                                 (short)0, c, false, false);
}

// ---------------------------------------------------------------------------
// Pass 1: fp32 -> bf16 conversion, 8 elems/thread-iter (b128 in, b128 out)
// ---------------------------------------------------------------------------
__global__ void cvt_bf16_kernel(const float* __restrict__ src,
                                unsigned short* __restrict__ dst, size_t n8) {
  size_t i = (size_t)blockIdx.x * blockDim.x + threadIdx.x;
  size_t stride = (size_t)gridDim.x * blockDim.x;
  const float4* s4 = reinterpret_cast<const float4*>(src);
  uint4* d4 = reinterpret_cast<uint4*>(dst);
  for (; i < n8; i += stride) {
    float4 a = s4[2 * i], b = s4[2 * i + 1];
    uint4 o;
    o.x = (unsigned)f2bf(a.x) | ((unsigned)f2bf(a.y) << 16);
    o.y = (unsigned)f2bf(a.z) | ((unsigned)f2bf(a.w) << 16);
    o.z = (unsigned)f2bf(b.x) | ((unsigned)f2bf(b.y) << 16);
    o.w = (unsigned)f2bf(b.z) | ((unsigned)f2bf(b.w) << 16);
    d4[i] = o;
  }
}

// ---------------------------------------------------------------------------
// Pass 2: V [B,S,D] fp32 -> V^T [B,D,S] bf16 (32x32 LDS tile transpose)
// ---------------------------------------------------------------------------
__global__ void vtrans_kernel(const float* __restrict__ V,
                              unsigned short* __restrict__ Vt) {
  __shared__ float t[32][33];
  const int b  = blockIdx.z;
  const int s0 = blockIdx.x * 32, d0 = blockIdx.y * 32;
  const int tx = threadIdx.x & 31, ty = threadIdx.x >> 5;
  for (int i = 0; i < 4; ++i) {
    int s = ty + i * 8;
    t[s][tx] = V[((size_t)b * S_ + s0 + s) * D_ + d0 + tx];
  }
  __syncthreads();
  for (int i = 0; i < 4; ++i) {
    int d = ty + i * 8;
    Vt[((size_t)b * D_ + d0 + d) * S_ + s0 + tx] = f2bf(t[tx][d]);
  }
}

// ---------------------------------------------------------------------------
// Pass 3: flash attention + Q residual.
// Block = 16 query rows x 8 waves; wave w owns D columns [w*128, w*128+128).
// ---------------------------------------------------------------------------
__global__ void __launch_bounds__(256)
attn_kernel(const unsigned short* __restrict__ Qb,
            const unsigned short* __restrict__ Kb,
            const unsigned short* __restrict__ Vt,
            const float* __restrict__ Qf,
            float* __restrict__ X,
            unsigned short* __restrict__ Xb) {
  __shared__ float part[NW][16][CK];          // per-wave partial scores
  __shared__ unsigned short Pt[16 * PST];     // softmaxed P tile (bf16, A layout)
  __shared__ float mrow[16], lrow[16], aRow[16];

  const int tid  = threadIdx.x;
  const int wave = tid >> 5, lane = tid & 31;
  const int hi   = lane >> 4, ln = lane & 15;
  const int b    = blockIdx.y;
  const int qb   = blockIdx.x * 16;

  const float inv_scale = 1.0f / (sqrtf((float)D_) + 1e-8f);

  if (wave == 0 && lane < 16) { mrow[lane] = -INFINITY; lrow[lane] = 0.0f; }

  // Preload this wave's Q A-tiles (16 rows x 128 d = 4 wmma A operands)
  v16bf qa[4];
  {
    const unsigned short* qrow =
        Qb + ((size_t)b * S_ + qb + ln) * D_ + wave * DSL;
    for (int s = 0; s < 4; ++s) qa[s] = loadA16(qrow + s * 32, hi);
  }

  v8f acc[8];
  for (int t = 0; t < 8; ++t)
    for (int v = 0; v < 8; ++v) acc[t][v] = 0.0f;

  __syncthreads();

  for (int jb = 0; jb < S_; jb += CK) {
    // ---- phase 1: partial scores over this wave's 128-wide D slice --------
    const unsigned short* k0 =
        Kb + ((size_t)b * S_ + jb + ln) * D_ + wave * DSL;
    const unsigned short* k1 = k0 + (size_t)16 * D_;
    if (jb + CK < S_) {
      __builtin_prefetch(k0 + (size_t)CK * D_, 0, 1);
      __builtin_prefetch(Vt + ((size_t)b * D_ + wave * DSL + ln) * S_ + jb + CK, 0, 1);
    }
    // batch ALL B-operand loads first so they clause & overlap the WMMAs
    v16bf kb[8];
    for (int s = 0; s < 4; ++s) {
      kb[s]     = loadB16(k0 + s * 32, hi);
      kb[4 + s] = loadB16(k1 + s * 32, hi);
    }
    SCHED_FENCE();                          // loads stay above, WMMAs below
    v8f c0, c1;
    for (int v = 0; v < 8; ++v) { c0[v] = 0.0f; c1[v] = 0.0f; }
    for (int s = 0; s < 4; ++s) {
      c0 = wmma_bf16(qa[s], kb[s],     c0);
      c1 = wmma_bf16(qa[s], kb[4 + s], c1);
    }
    for (int v = 0; v < 8; ++v) {            // C layout: row = v + 8*hi, col = ln
      part[wave][v + 8 * hi][ln]      = c0[v];
      part[wave][v + 8 * hi][16 + ln] = c1[v];
    }
    __syncthreads();

    // ---- phase 2: reduce partials + online softmax (wave w owns rows 2w,2w+1)
    {
      const int r = 2 * wave + hi;
      const int c = 2 * ln;
      float s0 = 0.0f, s1 = 0.0f;
      for (int p = 0; p < NW; ++p) {
        float2 pv = *reinterpret_cast<const float2*>(&part[p][r][c]);
        s0 += pv.x; s1 += pv.y;
      }
      s0 *= inv_scale; s1 *= inv_scale;
      float mx = fmaxf(s0, s1);
      for (int m = 1; m < 16; m <<= 1) mx = fmaxf(mx, __shfl_xor(mx, m, 32));
      const float mold = mrow[r];
      const float mnew = fmaxf(mold, mx);
      const float al   = __expf(mold - mnew);
      const float p0   = __expf(s0 - mnew);
      const float p1   = __expf(s1 - mnew);
      float ps = p0 + p1;
      for (int m = 1; m < 16; m <<= 1) ps += __shfl_xor(ps, m, 32);
      if (ln == 0) {
        mrow[r] = mnew;
        lrow[r] = lrow[r] * al + ps;
        aRow[r] = al;
      }
      unsigned pk = (unsigned)f2bf(p0) | ((unsigned)f2bf(p1) << 16);
      *reinterpret_cast<unsigned*>(&Pt[r * PST + 2 * ln]) = pk;
    }
    __syncthreads();

    // ---- phase 3: rescale accumulators, accumulate P @ V ------------------
    v16bf pa = loadA16(&Pt[ln * PST], hi);
    // batch V B-operand loads before the WMMA burst
    v16bf vbt[8];
    {
      const unsigned short* vb =
          Vt + ((size_t)b * D_ + wave * DSL + ln) * S_ + jb;
      for (int t = 0; t < 8; ++t)
        vbt[t] = loadB16(vb + (size_t)t * 16 * S_, hi);
    }
    for (int v = 0; v < 8; ++v) {
      const float al = aRow[v + 8 * hi];
      for (int t = 0; t < 8; ++t) acc[t][v] *= al;
    }
    SCHED_FENCE();                          // keep load batch ahead of WMMAs
    for (int t = 0; t < 8; ++t)
      acc[t] = wmma_bf16(pa, vbt[t], acc[t]);
    // (next iteration's post-phase-1 barrier orders LDS reuse)
  }

  // ---- epilogue: normalize, add Q residual, emit X (fp32 + bf16) ----------
  for (int v = 0; v < 8; ++v) {
    const float li  = 1.0f / lrow[v + 8 * hi];
    const int   row = qb + v + 8 * hi;
    for (int t = 0; t < 8; ++t) {
      const int col = wave * DSL + t * 16 + ln;
      const size_t idx = ((size_t)b * S_ + row) * D_ + col;
      const float val = acc[t][v] * li + Qf[idx];
      X[idx]  = val;
      Xb[idx] = f2bf(val);
    }
  }
}

// ---------------------------------------------------------------------------
// Pass 4/5: FFN GEMM  out[M,D] = act(A[M,D] @ W^T + bias) (+resid)
// One wave per 16x16 tile. The 16-token A panel (shared by all 8 waves of a
// block) is staged once into LDS via the gfx1250 async-to-LDS path.
// ---------------------------------------------------------------------------
__global__ void __launch_bounds__(256)
ffn_kernel(const unsigned short* __restrict__ Ain,   // [M][D] bf16
           const unsigned short* __restrict__ Wb,    // [N][D] bf16
           const float* __restrict__ bias,           // [D]
           const float* __restrict__ resid,          // [M][D] or null
           float* __restrict__ outF,                 // fp32 out or null
           unsigned short* __restrict__ outH,        // bf16 out or null
           int relu) {
  __shared__ unsigned short As[16 * AST];       // 16 x 1024 bf16, padded rows

  const int tid  = threadIdx.x;
  const int wave = tid >> 5, lane = tid & 31;
  const int hi   = lane >> 4, ln = lane & 15;
  const int rt   = blockIdx.x;                  // 16-token row tile
  const int ct   = blockIdx.y * NW + wave;      // 16-feature col tile

  // ---- stage A panel: 16 rows x 1024 bf16 = 2048 x b128 -------------------
  {
    const unsigned short* Ag = Ain + (size_t)rt * 16 * D_;
    for (int i = tid; i < 16 * 128; i += 256) {
      const int r = i >> 7, cq = i & 127;       // row, 8-elem chunk
      unsigned short* src = const_cast<unsigned short*>(Ag + (size_t)r * D_ + cq * 8);
      unsigned short* dst = &As[r * AST + cq * 8];
#if USE_ASYNC_LDS
      __builtin_amdgcn_global_load_async_to_lds_b128((gvi4_p)src, (lvi4_p)dst, 0, 0);
#else
      *reinterpret_cast<uint4*>(dst) = *reinterpret_cast<const uint4*>(src);
#endif
    }
#if USE_ASYNC_LDS
    __builtin_amdgcn_s_wait_asynccnt(0);
#endif
  }
  __syncthreads();

  v8f acc;
  for (int v = 0; v < 8; ++v) acc[v] = 0.0f;

  const unsigned short* aLds = &As[ln * AST];
  const unsigned short* bptr = Wb + ((size_t)ct * 16 + ln) * D_;
  for (int k = 0; k < D_; k += 128) {
    v16bf a0 = loadA16(aLds + k,      hi);
    v16bf a1 = loadA16(aLds + k + 32, hi);
    v16bf a2 = loadA16(aLds + k + 64, hi);
    v16bf a3 = loadA16(aLds + k + 96, hi);
    v16bf b0 = loadB16(bptr + k,      hi);
    v16bf b1 = loadB16(bptr + k + 32, hi);
    v16bf b2 = loadB16(bptr + k + 64, hi);
    v16bf b3 = loadB16(bptr + k + 96, hi);
    SCHED_FENCE();                          // keep load batch ahead of WMMAs
    acc = wmma_bf16(a0, b0, acc);
    acc = wmma_bf16(a1, b1, acc);
    acc = wmma_bf16(a2, b2, acc);
    acc = wmma_bf16(a3, b3, acc);
  }

  const int col = ct * 16 + ln;
  const float bv = bias[col];
  for (int v = 0; v < 8; ++v) {
    const int row = rt * 16 + v + 8 * hi;
    const size_t idx = (size_t)row * D_ + col;
    float val = acc[v] + bv;
    if (relu)  val = fmaxf(val, 0.0f);
    if (resid) val += resid[idx];
    if (outF)  outF[idx] = val;
    if (outH)  outH[idx] = f2bf(val);
  }
}

// ---------------------------------------------------------------------------
extern "C" void kernel_launch(void* const* d_in, const int* in_sizes, int n_in,
                              void* d_out, int out_size, void* d_ws, size_t ws_size,
                              hipStream_t stream) {
  const float* Q  = (const float*)d_in[0];
  const float* K  = (const float*)d_in[1];
  const float* V  = (const float*)d_in[2];
  const float* W1 = (const float*)d_in[3];
  const float* b1 = (const float*)d_in[4];
  const float* W2 = (const float*)d_in[5];
  const float* b2 = (const float*)d_in[6];
  float* out = (float*)d_out;

  const size_t nBSD = (size_t)B_ * S_ * D_;   // 33,554,432
  const size_t nDD  = (size_t)D_ * D_;        //  1,048,576

  // workspace layout (total ~452 MB, all 16B-aligned)
  unsigned short* Qb  = (unsigned short*)d_ws;
  unsigned short* Kb  = Qb  + nBSD;
  unsigned short* Vt  = Kb  + nBSD;
  unsigned short* W1b = Vt  + nBSD;
  unsigned short* W2b = W1b + nDD;
  unsigned short* Xb  = W2b + nDD;
  unsigned short* Hb  = Xb  + nBSD;
  float*          X   = (float*)(Hb + nBSD);

  // pass 1: conversions (n is in 8-element groups)
  cvt_bf16_kernel<<<2048, 256, 0, stream>>>(Q,  Qb,  nBSD / 8);
  cvt_bf16_kernel<<<2048, 256, 0, stream>>>(K,  Kb,  nBSD / 8);
  cvt_bf16_kernel<<<256,  256, 0, stream>>>(W1, W1b, nDD / 8);
  cvt_bf16_kernel<<<256,  256, 0, stream>>>(W2, W2b, nDD / 8);

  // pass 2: V transpose -> bf16
  vtrans_kernel<<<dim3(S_ / 32, D_ / 32, B_), 256, 0, stream>>>(V, Vt);

  // pass 3: flash attention + residual
  attn_kernel<<<dim3(S_ / 16, B_), 256, 0, stream>>>(Qb, Kb, Vt, Q, X, Xb);

  // pass 4: h = relu(X @ W1^T + b1)   (bf16 out for next GEMM)
  ffn_kernel<<<dim3((B_ * S_) / 16, D_ / 16 / NW), 256, 0, stream>>>(
      Xb, W1b, b1, nullptr, nullptr, Hb, 1);

  // pass 5: out = h @ W2^T + b2 + X   (fp32 out)
  ffn_kernel<<<dim3((B_ * S_) / 16, D_ / 16 / NW), 256, 0, stream>>>(
      Hb, W2b, b2, X, out, nullptr, 0);
}
